// SpaMamba_26319559590730
// MI455X (gfx1250) — matrix operand order repaired and checked by reading, not compiled, not run on hardware
//
#include <hip/hip_runtime.h>
#include <hip/hip_bf16.h>

typedef __attribute__((ext_vector_type(16))) _Float16 v16h;
typedef __attribute__((ext_vector_type(8)))  float    v8f;

// Problem constants (fixed by the reference)
#define BATCH   4
#define LSEQ    4096          // 64*64
#define DMODEL  128
#define DINNER  256
#define DSTATE  16
#define DTRANK  8
#define NXDB    40            // DTRANK + 2*DSTATE
#define NCHUNK  64
#define CHLEN   64            // LSEQ / NCHUNK
#define MTOT    (BATCH*LSEQ)  // 16384

// ---------------------------------------------------------------------------
// K0: (B,C,H*W) f32  ->  (B,H*W,C) f16   (LDS-tiled transpose)
// ---------------------------------------------------------------------------
__global__ __launch_bounds__(256) void pack_kernel(const float* __restrict__ x,
                                                   _Float16* __restrict__ xs) {
  __shared__ float t[32][33];
  const int b = blockIdx.z, l0 = blockIdx.x * 32, c0 = blockIdx.y * 32;
  for (int i = threadIdx.y; i < 32; i += 8)
    t[i][threadIdx.x] = x[((size_t)(b * DMODEL + c0 + i)) * LSEQ + l0 + threadIdx.x];
  __syncthreads();
  for (int i = threadIdx.y; i < 32; i += 8)
    xs[((size_t)b * LSEQ + l0 + i) * DMODEL + c0 + threadIdx.x] = (_Float16)t[threadIdx.x][i];
}

// ---------------------------------------------------------------------------
// WMMA GEMM:  C[m,n] (+)= sum_k A[src(m),k] * Bw[k,n]
//   A: f16 row-major [M,K]; staged to LDS via GLOBAL_LOAD_ASYNC_TO_LDS_B128
//     (row gather handles sequence reversal; addresses precomputed, advanced
//      by 64B per K-step so the hot loop has no address math)
//   Bw: f32 [K,N], converted to f16 while staging in LDS (transposed)
//   block tile 128x64, BK=32, 8 waves (4x2), each wave 2x2 x (16x16) frags
// ---------------------------------------------------------------------------
#define GBM 128
#define GBN 64
#define GBK 32

__global__ __launch_bounds__(256)
void gemm_wmma_kernel(const _Float16* __restrict__ A, const float* __restrict__ Bw,
                      float* __restrict__ C, int K, int N, int reverseRows, int accumulate) {
  __shared__ __align__(16) _Float16 lA[GBM * GBK];      // [row][k]
  __shared__ __align__(16) _Float16 lB[GBN * GBK];      // transposed: [n][k]
  const int tid  = threadIdx.x;
  const int lane = tid & 31, wave = tid >> 5;
  const int wm = wave >> 1, wn = wave & 1;              // 4 x 2 wave grid
  const int bm0 = blockIdx.y * GBM, bn0 = blockIdx.x * GBN;
  const int half = lane >> 4, lr = lane & 15;

  // --- per-thread async A-staging addresses (2 x 16B chunks per thread) ---
  // chunk c in {tid, tid+256}: row = c>>2 (0..127), seg = c&3 (8 halfs each)
  uint64_t gA0, gA1;
  uint32_t la0, la1;
  {
    int c0 = tid, c1 = tid + 256;
    int r0 = c0 >> 2, s0 = c0 & 3;
    int r1 = c1 >> 2, s1 = c1 & 3;
    int m0 = bm0 + r0, m1 = bm0 + r1, src0 = m0, src1 = m1;
    if (reverseRows) {
      int b0 = m0 >> 12, l0 = m0 & (LSEQ - 1); src0 = (b0 << 12) + (LSEQ - 1 - l0);
      int b1 = m1 >> 12, l1 = m1 & (LSEQ - 1); src1 = (b1 << 12) + (LSEQ - 1 - l1);
    }
    gA0 = (uint64_t)(uintptr_t)(A + (size_t)src0 * K + s0 * 8);
    gA1 = (uint64_t)(uintptr_t)(A + (size_t)src1 * K + s1 * 8);
    la0 = (uint32_t)(uintptr_t)&lA[r0 * GBK + s0 * 8];
    la1 = (uint32_t)(uintptr_t)&lA[r1 * GBK + s1 * 8];
  }

  v8f acc[2][2] = {};

  for (int k0 = 0; k0 < K; k0 += GBK) {
    // --- stage A tile via async DMA into LDS (no VGPR bounce, ASYNCcnt) ---
    asm volatile("global_load_async_to_lds_b128 %0, %1, off"
                 :: "v"(la0), "v"(gA0) : "memory");
    asm volatile("global_load_async_to_lds_b128 %0, %1, off"
                 :: "v"(la1), "v"(gA1) : "memory");
    gA0 += GBK * 2;  // 64 bytes per K-step
    gA1 += GBK * 2;

    // --- stage B tile (f32 -> f16, transposed); both loads issued first ---
    {
      int c0 = tid, c1 = tid + 256;                  // 512 float4 chunks
      int kk0 = c0 >> 4, n40 = (c0 & 15) * 4;
      int kk1 = c1 >> 4, n41 = (c1 & 15) * 4;
      const float* bp0 = Bw + (size_t)(k0 + kk0) * N + bn0 + n40;
      const float* bp1 = Bw + (size_t)(k0 + kk1) * N + bn0 + n41;
      float4 w0 = *(const float4*)bp0;
      float4 w1 = *(const float4*)bp1;
      if (k0 + GBK < K) {
        __builtin_prefetch(bp0 + (size_t)GBK * N, 0, 1);   // global_prefetch
        __builtin_prefetch(bp1 + (size_t)GBK * N, 0, 1);
      }
      lB[(n40 + 0) * GBK + kk0] = (_Float16)w0.x;
      lB[(n40 + 1) * GBK + kk0] = (_Float16)w0.y;
      lB[(n40 + 2) * GBK + kk0] = (_Float16)w0.z;
      lB[(n40 + 3) * GBK + kk0] = (_Float16)w0.w;
      lB[(n41 + 0) * GBK + kk1] = (_Float16)w1.x;
      lB[(n41 + 1) * GBK + kk1] = (_Float16)w1.y;
      lB[(n41 + 2) * GBK + kk1] = (_Float16)w1.z;
      lB[(n41 + 3) * GBK + kk1] = (_Float16)w1.w;
    }

    asm volatile("s_wait_asynccnt 0" ::: "memory");  // A-tile DMA complete
    __syncthreads();

#pragma unroll
    for (int fm = 0; fm < 2; ++fm) {
      // A fragment 16x32: lane half h, row lr; elem e -> K = 16h + e
      v16h a;
      const _Float16* pa = &lA[(wm * 32 + fm * 16 + lr) * GBK];
#pragma unroll
      for (int e = 0; e < 8; ++e) { a[e] = pa[half * 16 + e]; a[8 + e] = pa[half * 16 + 8 + e]; }
#pragma unroll
      for (int fn = 0; fn < 2; ++fn) {
        // B fragment 32x16: lane half h, col lr; elem e -> K = 16h + e (contiguous in lB)
        v16h bf;
        const _Float16* pb = &lB[(wn * 32 + fn * 16 + lr) * GBK + half * 16];
#pragma unroll
        for (int e = 0; e < 16; ++e) bf[e] = pb[e];
        acc[fm][fn] = __builtin_amdgcn_wmma_f32_16x16x32_f16(
            false, a, false, bf, (short)0, acc[fm][fn], false, false);
      }
    }
    __syncthreads();
  }

  // epilogue: C/D layout — VGPR r: row = 8*half + r, col = lr
#pragma unroll
  for (int fm = 0; fm < 2; ++fm)
#pragma unroll
    for (int fn = 0; fn < 2; ++fn)
#pragma unroll
      for (int r = 0; r < 8; ++r) {
        int m = bm0 + wm * 32 + fm * 16 + half * 8 + r;
        int n = bn0 + wn * 32 + fn * 16 + lr;
        size_t o = (size_t)m * N + n;
        float v = acc[fm][fn][r];
        if (accumulate) v += C[o];
        C[o] = v;
      }
}

// ---------------------------------------------------------------------------
// K2: causal 2-tap depthwise conv + SiLU  (scan order)
// ---------------------------------------------------------------------------
__global__ __launch_bounds__(256)
void conv_silu_kernel(const float* __restrict__ xz, const float* __restrict__ cw,
                      const float* __restrict__ cb, float* __restrict__ xc,
                      _Float16* __restrict__ xch) {
  size_t i = (size_t)blockIdx.x * 256 + threadIdx.x;   // over MTOT*DINNER
  int d = (int)(i & (DINNER - 1));
  size_t bl = i >> 8;
  int l = (int)(bl & (LSEQ - 1));
  float xp = xz[bl * (2 * DINNER) + d];
  float xprev = (l == 0) ? 0.f : xz[(bl - 1) * (2 * DINNER) + d];
  float v = xprev * cw[d * 2 + 0] + xp * cw[d * 2 + 1] + cb[d];
  float s = v / (1.f + __expf(-v));
  xc[i] = s;
  xch[i] = (_Float16)s;
}

// ---------------------------------------------------------------------------
// K3: xdb = xc @ W_x   (N=40 skinny; one output per thread)
// ---------------------------------------------------------------------------
__global__ __launch_bounds__(256)
void xdb_kernel(const _Float16* __restrict__ xch, const float* __restrict__ Wx,
                float* __restrict__ xdb) {
  int i = blockIdx.x * 256 + threadIdx.x;              // MTOT*NXDB (exact)
  int n = i % NXDB, m = i / NXDB;
  const _Float16* a = xch + (size_t)m * DINNER;
  float s = 0.f;
#pragma unroll 8
  for (int k = 0; k < DINNER; ++k) s += (float)a[k] * Wx[k * NXDB + n];
  xdb[i] = s;
}

// ---------------------------------------------------------------------------
// K4: dt = softplus(dt_r @ W_dt + b_dt)
// ---------------------------------------------------------------------------
__global__ __launch_bounds__(256)
void dt_kernel(const float* __restrict__ xdb, const float* __restrict__ Wdt,
               const float* __restrict__ bdt, float* __restrict__ dt) {
  int i = blockIdx.x * 256 + threadIdx.x;              // MTOT*DINNER
  int d = i & (DINNER - 1), m = i >> 8;
  const float* r = xdb + (size_t)m * NXDB;
  float s = bdt[d];
#pragma unroll
  for (int k = 0; k < DTRANK; ++k) s += r[k] * Wdt[k * DINNER + d];
  dt[i] = (s > 20.f) ? s : log1pf(__expf(s));
}

// ---------------------------------------------------------------------------
// K5a: chunked scan pass 1 — per-chunk local state h and decay product P
//      block = 256 threads = all of DINNER; grid = BATCH*NCHUNK
// ---------------------------------------------------------------------------
__global__ __launch_bounds__(256)
void scan_stats_kernel(const float* __restrict__ dt, const float* __restrict__ xc,
                       const float* __restrict__ xdb, const float* __restrict__ A_log,
                       float* __restrict__ chkH, float* __restrict__ chkP) {
  __shared__ float sB[DSTATE];
  const int d = threadIdx.x;
  const int b = blockIdx.x >> 6, ch = blockIdx.x & (NCHUNK - 1);
  float Av[DSTATE], h[DSTATE], P[DSTATE];
#pragma unroll
  for (int s = 0; s < DSTATE; ++s) { Av[s] = -__expf(A_log[d * DSTATE + s]); h[s] = 0.f; P[s] = 1.f; }
  const size_t base = (size_t)b * LSEQ + (size_t)ch * CHLEN;
  for (int t = 0; t < CHLEN; ++t) {
    size_t bl = base + t;
    if (d < DSTATE) sB[d] = xdb[bl * NXDB + DTRANK + d];
    __syncthreads();
    float dtv = dt[bl * DINNER + d];
    float dx  = dtv * xc[bl * DINNER + d];
#pragma unroll
    for (int s = 0; s < DSTATE; ++s) {
      float dA = __expf(dtv * Av[s]);
      h[s] = dA * h[s] + dx * sB[s];
      P[s] *= dA;
    }
    __syncthreads();
  }
  size_t o = ((size_t)blockIdx.x * DINNER + d) * DSTATE;
#pragma unroll
  for (int s = 0; s < DSTATE; ++s) { chkH[o + s] = h[s]; chkP[o + s] = P[s]; }
}

// ---------------------------------------------------------------------------
// K5b: serial combine across the 64 chunks -> per-chunk starting state
// ---------------------------------------------------------------------------
__global__ __launch_bounds__(256)
void scan_combine_kernel(const float* __restrict__ chkH, const float* __restrict__ chkP,
                         float* __restrict__ Hstart) {
  const int d = threadIdx.x, b = blockIdx.x;
  float h[DSTATE];
#pragma unroll
  for (int s = 0; s < DSTATE; ++s) h[s] = 0.f;
  for (int ch = 0; ch < NCHUNK; ++ch) {
    size_t o = (((size_t)b * NCHUNK + ch) * DINNER + d) * DSTATE;
#pragma unroll
    for (int s = 0; s < DSTATE; ++s) {
      Hstart[o + s] = h[s];
      h[s] = chkP[o + s] * h[s] + chkH[o + s];
    }
  }
}

// ---------------------------------------------------------------------------
// K5c: chunked scan pass 2 — emit y_t = <h_t, C_t> with correct start state
// ---------------------------------------------------------------------------
__global__ __launch_bounds__(256)
void scan_emit_kernel(const float* __restrict__ dt, const float* __restrict__ xc,
                      const float* __restrict__ xdb, const float* __restrict__ A_log,
                      const float* __restrict__ Hstart, float* __restrict__ ys) {
  __shared__ float sB[DSTATE], sC[DSTATE];
  const int d = threadIdx.x;
  const int b = blockIdx.x >> 6, ch = blockIdx.x & (NCHUNK - 1);
  float Av[DSTATE], h[DSTATE];
  size_t o = ((size_t)blockIdx.x * DINNER + d) * DSTATE;
#pragma unroll
  for (int s = 0; s < DSTATE; ++s) { Av[s] = -__expf(A_log[d * DSTATE + s]); h[s] = Hstart[o + s]; }
  const size_t base = (size_t)b * LSEQ + (size_t)ch * CHLEN;
  for (int t = 0; t < CHLEN; ++t) {
    size_t bl = base + t;
    if (d < DSTATE) { sB[d] = xdb[bl * NXDB + DTRANK + d]; sC[d] = xdb[bl * NXDB + DTRANK + DSTATE + d]; }
    __syncthreads();
    float dtv = dt[bl * DINNER + d];
    float dx  = dtv * xc[bl * DINNER + d];
    float y = 0.f;
#pragma unroll
    for (int s = 0; s < DSTATE; ++s) {
      float dA = __expf(dtv * Av[s]);
      h[s] = dA * h[s] + dx * sB[s];
      y += h[s] * sC[s];
    }
    ys[bl * DINNER + d] = y;
    __syncthreads();
  }
}

// ---------------------------------------------------------------------------
// K6: gate: g = (ys + D*xc) * silu(z); store f16 in ORIGINAL sequence order
// ---------------------------------------------------------------------------
__global__ __launch_bounds__(256)
void gate_kernel(const float* __restrict__ ys, const float* __restrict__ xc,
                 const float* __restrict__ xz, const float* __restrict__ Dp,
                 _Float16* __restrict__ gh, int reverse) {
  size_t i = (size_t)blockIdx.x * 256 + threadIdx.x;   // MTOT*DINNER
  int d = (int)(i & (DINNER - 1));
  size_t bl = i >> 8;
  int b = (int)(bl >> 12), l = (int)(bl & (LSEQ - 1));
  float z = xz[bl * (2 * DINNER) + DINNER + d];
  float g = (ys[i] + Dp[d] * xc[i]) * (z / (1.f + __expf(-z)));
  int lo = reverse ? (LSEQ - 1 - l) : l;
  gh[(((size_t)b << 12) + lo) * DINNER + d] = (_Float16)g;
}

// ---------------------------------------------------------------------------
// K7: GroupNorm(4 groups) + SiLU + residual, with (B,L,C)->(B,C,H,W) transpose
// ---------------------------------------------------------------------------
__global__ __launch_bounds__(256)
void gn_kernel(const float* __restrict__ ysum, const float* __restrict__ x,
               const float* __restrict__ gamma, const float* __restrict__ beta,
               float* __restrict__ out) {
  __shared__ float red[256];
  __shared__ float stats[2];
  const int b = blockIdx.x >> 2, g = blockIdx.x & 3;
  const int tid = threadIdx.x;
  const int CPG = DMODEL / 4;                 // 32 channels per group
  const int NEL = CPG * LSEQ;                 // 131072 elements per group
  float sum = 0.f, sq = 0.f;
  for (int i = tid; i < NEL; i += 256) {
    int c = g * CPG + (i >> 12), l = i & (LSEQ - 1);
    float v = ysum[(((size_t)b << 12) + l) * DMODEL + c];
    sum += v; sq += v * v;
  }
  red[tid] = sum; __syncthreads();
  for (int st = 128; st > 0; st >>= 1) { if (tid < st) red[tid] += red[tid + st]; __syncthreads(); }
  if (tid == 0) stats[0] = red[0] * (1.f / NEL);
  __syncthreads();
  red[tid] = sq; __syncthreads();
  for (int st = 128; st > 0; st >>= 1) { if (tid < st) red[tid] += red[tid + st]; __syncthreads(); }
  if (tid == 0) { float mu = stats[0]; stats[1] = rsqrtf(red[0] * (1.f / NEL) - mu * mu + 1e-5f); }
  __syncthreads();
  const float mu = stats[0], rinv = stats[1];
  for (int i = tid; i < NEL; i += 256) {
    int c = g * CPG + (i >> 12), l = i & (LSEQ - 1);
    float v = ysum[(((size_t)b << 12) + l) * DMODEL + c];
    float xn = (v - mu) * rinv * gamma[c] + beta[c];
    float s = xn / (1.f + __expf(-xn));
    size_t oi = (((size_t)(b * DMODEL + c)) << 12) + l;
    out[oi] = s + x[oi];
  }
}

// ---------------------------------------------------------------------------
// Host launcher
// ---------------------------------------------------------------------------
extern "C" void kernel_launch(void* const* d_in, const int* in_sizes, int n_in,
                              void* d_out, int out_size, void* d_ws, size_t ws_size,
                              hipStream_t stream) {
  (void)in_sizes; (void)n_in; (void)out_size; (void)ws_size;
  const float* x = (const float*)d_in[0];
  const float* gn_gamma = (const float*)d_in[19];
  const float* gn_beta  = (const float*)d_in[20];

  // workspace carve-up (256B aligned)
  char* ws = (char*)d_ws;
  size_t off = 0;
  auto carve = [&](size_t bytes) { void* p = ws + off; off = (off + bytes + 255) & ~(size_t)255; return p; };
  _Float16* xs_h  = (_Float16*)carve((size_t)MTOT * DMODEL * 2);
  float*    y_sum = (float*)   carve((size_t)MTOT * DMODEL * 4);
  float*    xz    = (float*)   carve((size_t)MTOT * 2 * DINNER * 4);
  float*    xc    = (float*)   carve((size_t)MTOT * DINNER * 4);
  _Float16* xc_h  = (_Float16*)carve((size_t)MTOT * DINNER * 2);
  float*    xdb   = (float*)   carve((size_t)MTOT * NXDB * 4);
  float*    dtb   = (float*)   carve((size_t)MTOT * DINNER * 4);
  float*    ysb   = (float*)   carve((size_t)MTOT * DINNER * 4);
  _Float16* g_h   = (_Float16*)carve((size_t)MTOT * DINNER * 2);
  float*    chkH  = (float*)   carve((size_t)BATCH * NCHUNK * DINNER * DSTATE * 4);
  float*    chkP  = (float*)   carve((size_t)BATCH * NCHUNK * DINNER * DSTATE * 4);
  float*    Hst   = (float*)   carve((size_t)BATCH * NCHUNK * DINNER * DSTATE * 4);

  // K0: pack/transpose + f16 convert
  pack_kernel<<<dim3(LSEQ / 32, DMODEL / 32, BATCH), dim3(32, 8), 0, stream>>>(x, xs_h);

  const int elemGrid = (MTOT * DINNER) / 256;          // 16384

  for (int dir = 0; dir < 2; ++dir) {
    const int base = 1 + 9 * dir;
    const float* W_in   = (const float*)d_in[base + 0];
    const float* conv_w = (const float*)d_in[base + 1];
    const float* conv_b = (const float*)d_in[base + 2];
    const float* W_x    = (const float*)d_in[base + 3];
    const float* W_dt   = (const float*)d_in[base + 4];
    const float* b_dt   = (const float*)d_in[base + 5];
    const float* A_log  = (const float*)d_in[base + 6];
    const float* Dp     = (const float*)d_in[base + 7];
    const float* W_out  = (const float*)d_in[base + 8];

    // xz = xs(reversed for dir=1) @ W_in  [16384 x 512 x 128]
    gemm_wmma_kernel<<<dim3((2 * DINNER) / GBN, MTOT / GBM), 256, 0, stream>>>(
        xs_h, W_in, xz, DMODEL, 2 * DINNER, dir, 0);

    conv_silu_kernel<<<elemGrid, 256, 0, stream>>>(xz, conv_w, conv_b, xc, xc_h);

    xdb_kernel<<<(MTOT * NXDB) / 256, 256, 0, stream>>>(xc_h, W_x, xdb);

    dt_kernel<<<elemGrid, 256, 0, stream>>>(xdb, W_dt, b_dt, dtb);

    // 3-pass chunked selective scan
    scan_stats_kernel<<<BATCH * NCHUNK, DINNER, 0, stream>>>(dtb, xc, xdb, A_log, chkH, chkP);
    scan_combine_kernel<<<BATCH, DINNER, 0, stream>>>(chkH, chkP, Hst);
    scan_emit_kernel<<<BATCH * NCHUNK, DINNER, 0, stream>>>(dtb, xc, xdb, A_log, Hst, ysb);

    // gate (+ un-reverse for dir=1)
    gate_kernel<<<elemGrid, 256, 0, stream>>>(ysb, xc, xz, Dp, g_h, dir);

    // y_sum (+)= g @ W_out  [16384 x 128 x 256]
    gemm_wmma_kernel<<<dim3(DMODEL / GBN, MTOT / GBM), 256, 0, stream>>>(
        g_h, W_out, y_sum, DINNER, DMODEL, 0, dir);
  }

  // GroupNorm + SiLU + residual, back to (B,C,H,W)
  gn_kernel<<<BATCH * 4, 256, 0, stream>>>(y_sum, x, gn_gamma, gn_beta, (float*)d_out);
}
